// RNNLayer_3298534884279
// MI455X (gfx1250) — matrix operand order repaired and checked by reading, not compiled, not run on hardware
//
#include <hip/hip_runtime.h>
#include <math.h>

typedef __attribute__((ext_vector_type(16))) __bf16 v16bf;
typedef __attribute__((ext_vector_type(8)))  __bf16 v8bf;
typedef __attribute__((ext_vector_type(8)))  float  v8f;
typedef __attribute__((ext_vector_type(4)))  unsigned int v4u;
typedef __attribute__((ext_vector_type(8)))  int v8i;
typedef __attribute__((ext_vector_type(4)))  int v4i;

#define BB 32
#define TT 2048
#define DD 256
#define HH 256
#define NG 768   // 3*H

#if __has_builtin(__builtin_amdgcn_tensor_load_to_lds)
#define HAVE_TDM 1
#if __has_include(<hip/amd_detail/amd_gfx1250_TDM.h>)
#define TDM_6ARG 1   // therock headers present -> 6-arg builtin arity
#endif
#endif

__device__ __forceinline__ float sigmoidf_(float x) {
    return 1.0f / (1.0f + __expf(-x));
}

// ---------------------------------------------------------------------------
// Kernel 0a: swizzle a 256x768 f32 weight matrix into bf16 WMMA B-fragment
// order: frag (ct,k) for column-tile ct (16 cols) and K-chunk k (32 rows);
// per lane 16 contiguous bf16 (32 B). One thread per (frag,lane).
// ---------------------------------------------------------------------------
__global__ __launch_bounds__(256) void wswz_kernel(const float* __restrict__ W,
                                                   __bf16* __restrict__ out) {
    const int idx   = blockIdx.x * 256 + threadIdx.x;    // 0..12287
    const int lane  = idx & 31;
    const int k     = (idx >> 5) & 7;
    const int ct    = idx >> 8;                          // 0..47
    const int l15   = lane & 15;
    const int lhalf = lane >> 4;
    const int kb    = k * 32 + lhalf * 8;
    const int col   = ct * 16 + l15;
    v16bf f;
#pragma unroll
    for (int i = 0; i < 16; ++i) f[i] = (__bf16)W[(kb + i + (i >= 8 ? 8 : 0)) * NG + col];
    *(v16bf*)&out[idx * 16] = f;
}

// Kernel 0b: X (f32) -> Xb (bf16), 8 elements per thread.
__global__ __launch_bounds__(256) void xcvt_kernel(const float* __restrict__ X,
                                                   __bf16* __restrict__ Xb) {
    const int t8 = (blockIdx.x * 256 + threadIdx.x) * 8;
    v8bf o;
#pragma unroll
    for (int i = 0; i < 8; ++i) o[i] = (__bf16)X[t8 + i];
    *(v8bf*)&Xb[t8] = o;
}

// ---------------------------------------------------------------------------
// Kernel 1: GX = Xb @ W_ih + b_ih (bf16 out). Wave = 64(M) x 16(N) strip.
// B-fragments and A-fragments are now straight b128 loads, no converts.
// ---------------------------------------------------------------------------
__global__ __launch_bounds__(256) void gx_gemm_kernel(
    const __bf16* __restrict__ Xb, const __bf16* __restrict__ WihB,
    const float* __restrict__ bih, __bf16* __restrict__ GX) {
    const int lane  = threadIdx.x & 31;
    const int l15   = lane & 15;
    const int lhalf = lane >> 4;
    const int wave  = blockIdx.x * 8 + (threadIdx.x >> 5);
    const int nt    = wave % 48;
    const int mt4   = wave / 48;
    const int rowBase = mt4 * 64;
    const int colBase = nt * 16;

    const v8f vzero = {0.f,0.f,0.f,0.f,0.f,0.f,0.f,0.f};
    v8f acc[4];
#pragma unroll
    for (int m = 0; m < 4; ++m) acc[m] = vzero;

#pragma unroll
    for (int k = 0; k < 8; ++k) {
        const v16bf bfrag = *(const v16bf*)&WihB[((nt * 8 + k) * 32 + lane) * 16];
        const int ka = k * 32 + lhalf * 8;
#pragma unroll
        for (int m = 0; m < 4; ++m) {
            const int row = rowBase + m * 16 + l15;
            const __bf16* xr = Xb + row * DD;
            const v8bf lo = *(const v8bf*)&xr[ka];
            const v8bf hi = *(const v8bf*)&xr[ka + 16];
            v16bf afrag;
#pragma unroll
            for (int i = 0; i < 8; ++i) { afrag[i] = lo[i]; afrag[i + 8] = hi[i]; }
            acc[m] = __builtin_amdgcn_wmma_f32_16x16x32_bf16(
                false, afrag, false, bfrag, (short)0, acc[m], false, false);
        }
    }

    const float bias = bih[colBase + l15];
#pragma unroll
    for (int m = 0; m < 4; ++m) {
#pragma unroll
        for (int i = 0; i < 8; ++i) {
            const int row = rowBase + m * 16 + lhalf * 8 + i;
            GX[row * NG + colBase + l15] = (__bf16)(acc[m][i] + bias);
        }
    }
}

// ---------------------------------------------------------------------------
// TDM: 2D tile load of 16 rows x 768 bf16 (rows strided T*3H elements)
// into LDS, descriptor per CDNA5 ISA.
// ---------------------------------------------------------------------------
#if HAVE_TDM
__device__ __forceinline__ void tdm_load_gx(const __bf16* gxrow, __bf16* ldsdst) {
    const unsigned long long ga = (unsigned long long)(size_t)(const void*)gxrow;
    const unsigned int la = (unsigned int)(size_t)(void*)ldsdst; // low 32b = LDS offset
    v4u g0;
    g0[0] = 1u;                                            // count=1, user mode
    g0[1] = la;                                            // lds_addr
    g0[2] = (unsigned int)ga;                              // global_addr[31:0]
    g0[3] = (unsigned int)((ga >> 32) & 0x1FFFFFFu) | 0x80000000u; // [56:32] | type=2
    v8i g1;
    g1[0] = 0x00010000;                // data_size=1 (2 bytes); no multicast
    g1[1] = (int)(768u << 16);         // tensor_dim0 = 768
    g1[2] = (int)(16u  << 16);         // tensor_dim1 = 16
    g1[3] = (int)(768u << 16);         // tile_dim0 = 768
    g1[4] = 16;                        // tile_dim1 = 16 (tile_dim2 = 0)
    g1[5] = TT * NG;                   // tensor_dim0_stride = 1572864
    g1[6] = 0;
    g1[7] = 0;
    const v4i gz = {0, 0, 0, 0};
#if TDM_6ARG
    const v8i gz8 = {0, 0, 0, 0, 0, 0, 0, 0};
    __builtin_amdgcn_tensor_load_to_lds(g0, g1, gz, gz, gz8, 0);
#else
    __builtin_amdgcn_tensor_load_to_lds(g0, g1, gz, gz, 0);
#endif
}
#endif

// ---------------------------------------------------------------------------
// Kernel 2: persistent GRU scan. 2 blocks x 16 waves; W_hh in VGPRs,
// h in LDS (bf16), GX tiles streamed by TDM into a double buffer.
// ---------------------------------------------------------------------------
__global__ __launch_bounds__(512) void gru_scan_kernel(
    const __bf16* __restrict__ GX, const __bf16* __restrict__ WhhB,
    const float* __restrict__ bhh, const float* __restrict__ h0,
    const unsigned char* __restrict__ mask, float* __restrict__ Y) {
    __shared__ __bf16 h_lds[16 * HH];                     // 8 KB
#if HAVE_TDM
    __shared__ __bf16 gx_lds[2][16 * NG];                 // 48 KB double buffer
#endif
    const int mt    = blockIdx.x;
    const int lane  = threadIdx.x & 31;
    const int l15   = lane & 15;
    const int lhalf = lane >> 4;
    const int w     = threadIdx.x >> 5;                   // column group 0..15
    const int colg  = w * 16 + l15;

    // W_hh B-fragments: straight 32-B loads from the pre-swizzled buffer.
    v16bf Bf[3][8];
#pragma unroll
    for (int g = 0; g < 3; ++g)
#pragma unroll
        for (int k = 0; k < 8; ++k)
            Bf[g][k] = *(const v16bf*)&WhhB[(((g * 16 + w) * 8 + k) * 32 + lane) * 16];

    const float bh_r = bhh[colg];
    const float bh_z = bhh[HH + colg];
    const float bh_n = bhh[2 * HH + colg];

    for (int idx = threadIdx.x; idx < 16 * HH; idx += 512) {
        const int r = idx >> 8, c = idx & 255;
        h_lds[idx] = (__bf16)h0[(mt * 16 + r) * HH + c];
    }

    const __bf16* gx_base = GX + (size_t)(mt * 16) * TT * NG; // (b0, t=0) row
#if HAVE_TDM
    if (w == 0) {
        tdm_load_gx(gx_base, &gx_lds[0][0]);              // t = 0 tile
        __builtin_amdgcn_s_wait_tensorcnt(0);
    }
#endif
    __syncthreads();

    const v8f vzero = {0.f,0.f,0.f,0.f,0.f,0.f,0.f,0.f};

    for (int t = 0; t < TT; ++t) {
        const int buf = t & 1;
#if HAVE_TDM
        if (w == 0 && t + 1 < TT)                         // prefetch t+1 tile
            tdm_load_gx(gx_base + (size_t)(t + 1) * NG, &gx_lds[buf ^ 1][0]);
#endif
        // ---- gh = h @ W_hh : 3 gate tiles, K = 8 x 32 --------------------
        v8f accr = vzero, accz = vzero, accn = vzero;
#pragma unroll
        for (int k = 0; k < 8; ++k) {
            const int off = l15 * HH + k * 32 + lhalf * 8;
            const v8bf lo = *(const v8bf*)&h_lds[off];
            const v8bf hi = *(const v8bf*)&h_lds[off + 16];
            v16bf a;
#pragma unroll
            for (int i = 0; i < 8; ++i) { a[i] = lo[i]; a[i + 8] = hi[i]; }
            accr = __builtin_amdgcn_wmma_f32_16x16x32_bf16(
                false, a, false, Bf[0][k], (short)0, accr, false, false);
            accz = __builtin_amdgcn_wmma_f32_16x16x32_bf16(
                false, a, false, Bf[1][k], (short)0, accz, false, false);
            accn = __builtin_amdgcn_wmma_f32_16x16x32_bf16(
                false, a, false, Bf[2][k], (short)0, accn, false, false);
        }

        // ---- fused gates + masked update ---------------------------------
        float hnv[8];
#pragma unroll
        for (int i = 0; i < 8; ++i) {
            const int bl = i + lhalf * 8;
            const int b  = mt * 16 + bl;
            const int grow = b * TT + t;
#if HAVE_TDM
            const float gxr = (float)gx_lds[buf][bl * NG + colg];
            const float gxz = (float)gx_lds[buf][bl * NG + HH + colg];
            const float gxn = (float)gx_lds[buf][bl * NG + 2 * HH + colg];
#else
            const float gxr = (float)GX[(size_t)grow * NG + colg];
            const float gxz = (float)GX[(size_t)grow * NG + HH + colg];
            const float gxn = (float)GX[(size_t)grow * NG + 2 * HH + colg];
#endif
            const float r = sigmoidf_(gxr + accr[i] + bh_r);
            const float z = sigmoidf_(gxz + accz[i] + bh_z);
            const float n = tanhf(gxn + r * (accn[i] + bh_n));
            const float hold = (float)h_lds[bl * HH + colg];
            float hnew = (1.0f - z) * n + z * hold;
            if (!mask[grow]) hnew = hold;
            hnv[i] = hnew;
            Y[(size_t)grow * HH + colg] = hnew;
        }

        __syncthreads();                       // all reads of h / gx done
#pragma unroll
        for (int i = 0; i < 8; ++i) {
            const int bl = i + lhalf * 8;
            h_lds[bl * HH + colg] = (__bf16)hnv[i];
        }
#if HAVE_TDM
        if (w == 0) __builtin_amdgcn_s_wait_tensorcnt(0);  // t+1 tile landed
#endif
        __syncthreads();                       // new h + gx tile visible
    }
}

// ---------------------------------------------------------------------------
extern "C" void kernel_launch(void* const* d_in, const int* in_sizes, int n_in,
                              void* d_out, int out_size, void* d_ws, size_t ws_size,
                              hipStream_t stream) {
    const float*         x    = (const float*)d_in[0];
    const unsigned char* mask = (const unsigned char*)d_in[1];
    const float*         h0   = (const float*)d_in[2];
    const float*         Wih  = (const float*)d_in[3];
    const float*         Whh  = (const float*)d_in[4];
    const float*         bih  = (const float*)d_in[5];
    const float*         bhh  = (const float*)d_in[6];
    float*               y    = (float*)d_out;

    __bf16* ws   = (__bf16*)d_ws;
    __bf16* GX   = ws;                          // 50,331,648 elems (96 MB)
    __bf16* Xb   = ws + 50331648;               // 16,777,216 elems (32 MB)
    __bf16* WihB = ws + 50331648 + 16777216;    // 196,608 elems
    __bf16* WhhB = WihB + 196608;               // 196,608 elems

    wswz_kernel<<<48, 256, 0, stream>>>(Wih, WihB);
    wswz_kernel<<<48, 256, 0, stream>>>(Whh, WhhB);
    xcvt_kernel<<<8192, 256, 0, stream>>>(x, Xb);
    gx_gemm_kernel<<<6144, 256, 0, stream>>>(Xb, WihB, bih, GX);
    gru_scan_kernel<<<2, 512, 0, stream>>>(GX, WhhB, bhh, h0, mask, y);
}
